// DescrptSeARho_32160715112591
// MI455X (gfx1250) — matrix operand order repaired and checked by reading, not compiled, not run on hardware
//
#include <hip/hip_runtime.h>
#include <stdint.h>

#define NLOC 8192
#define NNEI 120
#define NTYP 2

// d_ws layout in dwords
#define W2PACK_DW 0        // 2*4*32*8   = 2048 dwords
#define W3PACK_DW 2048     // 2*7*2*32*8 = 7168 dwords
#define ENV_DW    16384    // 8192 atoms * 128 slots * 4 = 4,194,304 dwords

// d_out layout (floats): result | rot_mat | sw
#define RES_OFF 0
#define ROT_OFF 13107200   // 8192*1600
#define SW_OFF  15564800   // ROT_OFF + 8192*300

typedef __attribute__((ext_vector_type(16))) _Float16 v16h;
typedef __attribute__((ext_vector_type(8)))  float    v8f;
typedef __attribute__((ext_vector_type(4)))  unsigned u32x4;
typedef __attribute__((ext_vector_type(4)))  float    f32x4;

typedef union { v16h h; u32x4 u[2]; } Frag16;
typedef union { _Float16 h[2]; unsigned u; } H2;

// ---------------------------------------------------------------------------
// Branchless hardware tanh (CDNA5 V_TANH_F32 transcendental).
// ---------------------------------------------------------------------------
__device__ __forceinline__ float fast_tanh(float x) {
#if __has_builtin(__builtin_amdgcn_tanhf)
  return __builtin_amdgcn_tanhf(x);
#else
  float r;
  asm("v_tanh_f32 %0, %1" : "=v"(r) : "v"(x));
  return r;
#endif
}

// ---------------------------------------------------------------------------
// Kernel 1: pack W2/W3 into WMMA B-operand layout (f16), zero-padded.
// B layout for v_wmma_f32_16x16x32_f16: lane holds column n=(lane&15);
// lane half h: dwords 0-3 hold K=8h..8h+7, dwords 4-7 hold K=16+8h..16+8h+7.
// ---------------------------------------------------------------------------
__global__ void pack_weights_kernel(const float* __restrict__ W2,
                                    const float* __restrict__ W3,
                                    unsigned* __restrict__ ws) {
  int gid = blockIdx.x * blockDim.x + threadIdx.x;
  if (gid < 2048) {
    // index = ((tt*4 + nt)*32 + lane)*8 + v
    int v    = gid & 7;
    int lane = (gid >> 3) & 31;
    int nt   = (gid >> 8) & 3;
    int tt   = gid >> 10;
    int n = nt * 16 + (lane & 15);
    int k = ((lane >> 4) & 1) * 8 + (v >> 2) * 16 + 2 * (v & 3);
    H2 p;
    p.h[0] = (k     < 25 && n < 50) ? (_Float16)W2[(tt * 25 + k    ) * 50 + n] : (_Float16)0.f;
    p.h[1] = (k + 1 < 25 && n < 50) ? (_Float16)W2[(tt * 25 + k + 1) * 50 + n] : (_Float16)0.f;
    ws[W2PACK_DW + gid] = p.u;
  } else if (gid < 2048 + 7168) {
    // index = (((tt*7 + nt)*2 + ks)*32 + lane)*8 + v
    int g2 = gid - 2048;
    int v    = g2 & 7;
    int t    = g2 >> 3;
    int lane = t & 31; t >>= 5;
    int ks   = t & 1;  t >>= 1;
    int nt   = t % 7;
    int tt   = t / 7;
    int n = nt * 16 + (lane & 15);
    int k = ks * 32 + ((lane >> 4) & 1) * 8 + (v >> 2) * 16 + 2 * (v & 3);
    H2 p;
    p.h[0] = (k     < 50 && n < 100) ? (_Float16)W3[(tt * 50 + k    ) * 100 + n] : (_Float16)0.f;
    p.h[1] = (k + 1 < 50 && n < 100) ? (_Float16)W3[(tt * 50 + k + 1) * 100 + n] : (_Float16)0.f;
    ws[W3PACK_DW + g2] = p.u;
  }
}

// ---------------------------------------------------------------------------
// Kernel 2: environment matrix + smooth weight.
// One thread per (atom, padded slot). Slots: [0..59]=nbr 0..59, [60..63]=pad0,
// [64..123]=nbr 60..119, [124..127]=pad0.
// ---------------------------------------------------------------------------
__global__ __launch_bounds__(256)
void env_mat_kernel(const float* __restrict__ coord,
                    const int*   __restrict__ atype,
                    const int*   __restrict__ nlist,
                    const float* __restrict__ davg,
                    const float* __restrict__ dstd,
                    float* __restrict__ envWS,
                    float* __restrict__ out) {
  int tid = threadIdx.x;
  int i  = blockIdx.x * 2 + (tid >> 7);
  int jj = tid & 127;
  float* eslot = envWS + ((size_t)i * 128 + jj) * 4;
  bool pad = (jj >= 60 && jj < 64) || (jj >= 124);
  if (pad) {
    f32x4 z; z.x = 0.f; z.y = 0.f; z.z = 0.f; z.w = 0.f;
    *(f32x4*)eslot = z;
    return;
  }
  int j = jj - (jj >= 64 ? 4 : 0);
  int at = atype[i];
  int nl = nlist[(size_t)i * NNEI + j];
  bool mask = nl >= 0;
  int idx = mask ? nl : 0;
  float dx = coord[idx * 3 + 0] - coord[i * 3 + 0];
  float dy = coord[idx * 3 + 1] - coord[i * 3 + 1];
  float dz = coord[idx * 3 + 2] - coord[i * 3 + 2];
  float r = sqrtf(dx * dx + dy * dy + dz * dz);
  r += mask ? 0.f : 1.f;
  float t0 = 1.f / r;
  float inv2 = t0 * t0;
  // smooth weight
  float uu = (r - 0.5f) * (1.f / 5.5f);
  float vv = uu * uu * uu * (uu * (-6.f * uu + 15.f) - 10.f) + 1.f;
  float sw = (r < 0.5f) ? 1.f : ((r > 6.f) ? 0.f : vv);
  sw *= mask ? 1.f : 0.f;
  float raw0 = t0 * sw;
  float raw1 = dx * inv2 * sw;
  float raw2 = dy * inv2 * sw;
  float raw3 = dz * inv2 * sw;
  const float* da = davg + ((size_t)at * NNEI + j) * 4;
  const float* ds = dstd + ((size_t)at * NNEI + j) * 4;
  f32x4 e;
  e.x = (raw0 - da[0]) / ds[0];
  e.y = (raw1 - da[1]) / ds[1];
  e.z = (raw2 - da[2]) / ds[2];
  e.w = (raw3 - da[3]) / ds[3];
  *(f32x4*)eslot = e;
  out[SW_OFF + (size_t)i * NNEI + j] = sw;
}

// ---------------------------------------------------------------------------
// Kernel 3: embedding MLP via WMMA + reductions. One wave per atom.
// ---------------------------------------------------------------------------
__device__ __forceinline__ void wave_lds_fence() {
  asm volatile("s_wait_dscnt 0x0" ::: "memory");
}

__device__ __forceinline__ v8f wmma_f16(const Frag16& a, const Frag16& b, v8f c) {
  return __builtin_amdgcn_wmma_f32_16x16x32_f16(false, a.h, false, b.h,
                                                (short)0, c, false, false);
}

__global__ __launch_bounds__(128)
void embed_kernel(const float* __restrict__ envWS,
                  const u32x4* __restrict__ w2p,
                  const u32x4* __restrict__ w3p,
                  const float* __restrict__ W1,
                  const float* __restrict__ B1,
                  const float* __restrict__ B2,
                  const float* __restrict__ B3,
                  float* __restrict__ out) {
  __shared__ __align__(16) float    s_env[4][128 * 4];  // 8 KB
  __shared__ __align__(16) _Float16 s_x1[4][64 * 32];   // 16 KB
  __shared__ __align__(16) _Float16 s_x2[4][64 * 64];   // 32 KB

  const int lane = threadIdx.x & 31;
  const int wv   = threadIdx.x >> 5;
  const int atom = blockIdx.x * 4 + wv;
  const int hib  = (lane & 16) ? 16 : 0;   // high-half lanes: +16B in K

  float*    envL = s_env[wv];
  _Float16* x1L  = s_x1[wv];
  _Float16* x2L  = s_x2[wv];

  // Load padded env tile: 128 slots x float4
  {
    const f32x4* ep = (const f32x4*)envWS + (size_t)atom * 128;
    f32x4* el = (f32x4*)envL;
#pragma unroll
    for (int r = 0; r < 4; ++r) {
      int s = lane + 32 * r;
      el[s] = ep[s];
    }
  }
  wave_lds_fence();

  float part[7][4];
#pragma unroll
  for (int nt = 0; nt < 7; ++nt)
#pragma unroll
    for (int d = 0; d < 4; ++d) part[nt][d] = 0.f;

  for (int tt = 0; tt < NTYP; ++tt) {
    // ---- Layer 1: x1 = tanh(ss*W1 + b1), 64 rows x 32 cols (cols>=25 zero) ----
#pragma unroll
    for (int rh = 0; rh < 2; ++rh) {
      int r = lane + 32 * rh;
      float ss = envL[(tt * 64 + r) * 4 + 0];
#pragma unroll
      for (int c = 0; c < 32; ++c) {
        float v = 0.f;
        if (c < 25) v = fast_tanh(ss * W1[tt * 25 + c] + B1[tt * 25 + c]);
        x1L[r * 32 + c] = (_Float16)v;
      }
    }
    wave_lds_fence();

    // ---- Layer 2: y = x1(64x32) @ W2p(32x64); x2 = tanh(y+b2)+concat(x1,x1) ----
    for (int nt = 0; nt < 4; ++nt) {
      Frag16 b;
      int bi = ((tt * 4 + nt) * 32 + lane) * 2;
      b.u[0] = w2p[bi];
      b.u[1] = w2p[bi + 1];
      int n = nt * 16 + (lane & 15);
      float b2v = (n < 50) ? B2[tt * 50 + n] : 0.f;
      for (int mt = 0; mt < 4; ++mt) {
        Frag16 a;
        int row = mt * 16 + (lane & 15);
        const char* abase = (const char*)x1L + row * 64 + hib;
        a.u[0] = *(const u32x4*)(abase);
        a.u[1] = *(const u32x4*)(abase + 32);
        v8f acc = {};
        acc = wmma_f16(a, b, acc);
        int rbase = mt * 16 + ((lane & 16) ? 8 : 0);
#pragma unroll
        for (int v = 0; v < 8; ++v) {
          int s = rbase + v;
          float val = 0.f;
          if (n < 50) {
            float skip = (float)x1L[s * 32 + (n < 25 ? n : n - 25)];
            val = fast_tanh(acc[v] + b2v) + skip;
          }
          x2L[s * 64 + n] = (_Float16)val;
        }
      }
    }
    wave_lds_fence();

    // ---- Layer 3: y = x2(64x64) @ W3p(64x112); gg = tanh(y+b3)+concat(x2,x2)
    //      then xyz[d][g] += sum_s rr[s][d]*gg[s][g] (rr pad rows are zero) ----
    for (int nt = 0; nt < 7; ++nt) {
      Frag16 bA, bB;
      int bi = ((tt * 7 + nt) * 2 * 32 + lane) * 2;   // ks = 0
      bA.u[0] = w3p[bi];
      bA.u[1] = w3p[bi + 1];
      bB.u[0] = w3p[bi + 64];                          // ks = 1
      bB.u[1] = w3p[bi + 65];
      int g = nt * 16 + (lane & 15);
      float b3v = (g < 100) ? B3[tt * 100 + g] : 0.f;
      for (int mt = 0; mt < 4; ++mt) {
        Frag16 a0, a1;
        int row = mt * 16 + (lane & 15);
        const char* abase = (const char*)x2L + row * 128 + hib;
        a0.u[0] = *(const u32x4*)(abase);
        a0.u[1] = *(const u32x4*)(abase + 32);
        a1.u[0] = *(const u32x4*)(abase + 64);
        a1.u[1] = *(const u32x4*)(abase + 96);
        v8f acc = {};
        acc = wmma_f16(a0, bA, acc);
        acc = wmma_f16(a1, bB, acc);
        int rbase = mt * 16 + ((lane & 16) ? 8 : 0);
#pragma unroll
        for (int v = 0; v < 8; ++v) {
          int s = rbase + v;
          float gg = 0.f;
          if (g < 100) {
            float skip = (float)x2L[s * 64 + (g < 50 ? g : g - 50)];
            gg = fast_tanh(acc[v] + b3v) + skip;
          }
          f32x4 e = ((const f32x4*)envL)[tt * 64 + s];
          part[nt][0] += gg * e.x;
          part[nt][1] += gg * e.y;
          part[nt][2] += gg * e.z;
          part[nt][3] += gg * e.w;
        }
      }
    }
    wave_lds_fence();
  }

  // ---- combine lane halves, scale by 1/nnei ----
  const float inv = 1.f / (float)NNEI;
#pragma unroll
  for (int nt = 0; nt < 7; ++nt)
#pragma unroll
    for (int d = 0; d < 4; ++d) {
      float v = part[nt][d];
      v += __shfl_xor(v, 16, 32);
      part[nt][d] = v * inv;
    }

  // ---- stage xyz[:, 0:16] (tile 0) for the Gram product ----
  float* xk = (float*)x1L;   // reuse x1 region: 64 floats
  if (lane < 16) {
#pragma unroll
    for (int d = 0; d < 4; ++d) xk[d * 16 + lane] = part[0][d];
  }
  wave_lds_fence();

  // ---- rot_mat + result ----
  if (lane < 16) {
    float* resO = out + RES_OFF + (size_t)atom * 1600;
    float* rotO = out + ROT_OFF + (size_t)atom * 300;
#pragma unroll
    for (int nt = 0; nt < 7; ++nt) {
      int g = nt * 16 + lane;
      if (g < 100) {
        rotO[g * 3 + 0] = part[nt][1];
        rotO[g * 3 + 1] = part[nt][2];
        rotO[g * 3 + 2] = part[nt][3];
#pragma unroll
        for (int k4 = 0; k4 < 4; ++k4) {
          float r0 = 0.f, r1 = 0.f, r2 = 0.f, r3 = 0.f;
#pragma unroll
          for (int d = 0; d < 4; ++d) {
            float pg = part[nt][d];
            r0 += pg * xk[d * 16 + k4 * 4 + 0];
            r1 += pg * xk[d * 16 + k4 * 4 + 1];
            r2 += pg * xk[d * 16 + k4 * 4 + 2];
            r3 += pg * xk[d * 16 + k4 * 4 + 3];
          }
          f32x4 rv; rv.x = r0; rv.y = r1; rv.z = r2; rv.w = r3;
          *(f32x4*)(resO + g * 16 + k4 * 4) = rv;
        }
      }
    }
  }
}

// ---------------------------------------------------------------------------
extern "C" void kernel_launch(void* const* d_in, const int* in_sizes, int n_in,
                              void* d_out, int out_size, void* d_ws, size_t ws_size,
                              hipStream_t stream) {
  const float* coord = (const float*)d_in[0];
  const int*   atype = (const int*)d_in[1];
  const int*   nlist = (const int*)d_in[2];
  const float* davg  = (const float*)d_in[3];
  const float* dstd  = (const float*)d_in[4];
  const float* W1    = (const float*)d_in[5];
  const float* b1    = (const float*)d_in[6];
  const float* W2    = (const float*)d_in[7];
  const float* b2    = (const float*)d_in[8];
  const float* W3    = (const float*)d_in[9];
  const float* b3    = (const float*)d_in[10];
  float*    out  = (float*)d_out;
  unsigned* wsu  = (unsigned*)d_ws;
  float*    envW = (float*)d_ws + ENV_DW;

  pack_weights_kernel<<<36, 256, 0, stream>>>(W2, W3, wsu);
  env_mat_kernel<<<NLOC / 2, 256, 0, stream>>>(coord, atype, nlist, davg, dstd, envW, out);
  embed_kernel<<<NLOC / 4, 128, 0, stream>>>(
      envW,
      (const u32x4*)(wsu + W2PACK_DW),
      (const u32x4*)(wsu + W3PACK_DW),
      W1, b1, b2, b3, out);
}
